// HardwareKANLayer_41875931136528
// MI455X (gfx1250) — matrix operand order repaired and checked by reading, not compiled, not run on hardware
//
#include <hip/hip_runtime.h>
#include <hip/hip_bf16.h>

// ---------------------------------------------------------------------------
// KAN layer on MI455X (gfx1250):
//   out = clip( silu(x) @ base_w^T + einsum('bik,oik->bo', bspline(x), spline_w), -1, 1 )
//
// Three phases (all launched on `stream`):
//   1) pack_weights : Wp[o][12288] bf16 in d_ws      (25 MB, L2-resident)
//   2) build_A      : Ax[b][12288] bf16 in d_ws      (192 MB) — silu + 11 bases,
//                     computed ONCE per element, fully coalesced (basis-major K)
//   3) kan_gemm     : plain bf16 WMMA GEMM 8192 x 1024 x 12288, fp32 accum
//
// K index order (shared by Ax and Wp):
//   r in [0,1024)        : silu region, r == i
//   r in [1024,12288)    : kk = (r-1024)>>10, i = (r-1024)&1023  (basis-major)
// ---------------------------------------------------------------------------

typedef __attribute__((ext_vector_type(16))) __bf16 v16bf;
typedef __attribute__((ext_vector_type(8)))  float  v8f;

#define BATCH 8192
#define INF   1024
#define OUTF  1024
#define NBAS  11
#define KTOT  (INF + INF * NBAS)   // 12288

#define BM 128
#define BN 256

__device__ __forceinline__ unsigned short f32_to_bf16(float f) {
    unsigned u = __float_as_uint(f);
    unsigned r = u + 0x7FFFu + ((u >> 16) & 1u);   // round-to-nearest-even
    return (unsigned short)(r >> 16);
}

// ---------------------------------------------------------------------------
// Phase 1: pack weights -> bf16, K reordered to match Ax.
// grid = (KTOT/256, OUTF)
// ---------------------------------------------------------------------------
__global__ void pack_weights_kernel(const float* __restrict__ bw,
                                    const float* __restrict__ sw,
                                    unsigned short* __restrict__ wp) {
    const int r = blockIdx.x * blockDim.x + threadIdx.x;   // 0..KTOT-1
    const int o = blockIdx.y;
    float v;
    if (r < INF) {
        v = bw[(size_t)o * INF + r];
    } else {
        const int kk = (r - INF) >> 10;        // basis index 0..10
        const int i  = (r - INF) & (INF - 1);  // input feature
        v = sw[((size_t)o * INF + i) * NBAS + kk];
    }
    wp[(size_t)o * KTOT + r] = f32_to_bf16(v);
}

// ---------------------------------------------------------------------------
// Phase 2: activation matrix. One thread per (b,i): 1 coalesced x load,
// 1 silu + 11 cubic B-spline bases, 12 coalesced bf16 stores.
// grid = (INF/256, BATCH)
// ---------------------------------------------------------------------------
__global__ void build_A_kernel(const float* __restrict__ x,
                               unsigned short* __restrict__ ax) {
    const int i = blockIdx.x * blockDim.x + threadIdx.x;   // 0..1023
    const int b = blockIdx.y;                              // 0..8191
    const float xv = x[(size_t)b * INF + i];

    unsigned short* row = ax + (size_t)b * KTOT;

    // silu region
    row[i] = f32_to_bf16(xv / (1.0f + __expf(-xv)));

    // cubic B-spline bases, centers -1.25 + 0.25*kk, u = |x-c|/0.25
#pragma unroll
    for (int kk = 0; kk < NBAS; ++kk) {
        const float c  = -1.25f + 0.25f * (float)kk;
        const float u  = fabsf(xv - c) * 4.0f;
        const float c2 = 2.0f - u;
        const float c1 = 1.0f - u;
        const float c23   = c2 * c2 * c2;
        const float inner = (1.0f / 6.0f) * (c23 - 4.0f * c1 * c1 * c1);
        const float outer = (1.0f / 6.0f) * c23;
        const float v = (u < 1.0f) ? inner : ((u < 2.0f) ? outer : 0.0f);
        row[INF + (kk << 10) + i] = f32_to_bf16(v);
    }
}

// ---------------------------------------------------------------------------
// Phase 3: bf16 WMMA GEMM, M=8192, N=1024, K=12288.
// block = 256 threads = 8 waves; block tile 128x256; wave tile 64x64.
// Fragments loaded directly from global (B in L2, A deduped via L0/L1).
// grid = (OUTF/BN, BATCH/BM) = (4, 64)
// ---------------------------------------------------------------------------
__global__ __launch_bounds__(256)
void kan_gemm_kernel(const unsigned short* __restrict__ ax,
                     const unsigned short* __restrict__ wp,
                     float* __restrict__ out) {
    const int tid  = threadIdx.x;
    const int lane = tid & 31;
    const int wave = tid >> 5;          // 0..7
    const int wm   = wave & 1;          // M offset = wm*64
    const int wn   = wave >> 1;         // N offset = wn*64
    const int nlo  = lane & 15;
    const int half = lane >> 4;

    const int bBase = blockIdx.y * BM + wm * 64;
    const int oBase = blockIdx.x * BN + wn * 64;

    // Per-lane base pointers (K advances by 32 each step).
    // A fragment (16-bit A 16x32): per lane K = {half*8..+7} U {16+half*8..+7}
    const unsigned short* aptr[4];
#pragma unroll
    for (int ms = 0; ms < 4; ++ms)
        aptr[ms] = ax + (size_t)(bBase + ms * 16 + nlo) * KTOT + half * 8;

    // B fragment (16-bit B 32x16): per lane K = {half*8..+7} U {16+half*8..+7}
    const unsigned short* bptr[4];
#pragma unroll
    for (int ns = 0; ns < 4; ++ns)
        bptr[ns] = wp + (size_t)(oBase + ns * 16 + nlo) * KTOT + half * 8;

    v8f acc[4][4];
#pragma unroll
    for (int ms = 0; ms < 4; ++ms)
#pragma unroll
        for (int ns = 0; ns < 4; ++ns)
            acc[ms][ns] = (v8f)0.0f;

    for (int kc = 0; kc < KTOT; kc += 32) {
        union frag { uint4 u[2]; v16bf v; };

        frag bf[4];
#pragma unroll
        for (int ns = 0; ns < 4; ++ns) {
            const unsigned short* p = bptr[ns] + kc;
            bf[ns].u[0] = *(const uint4*)(p);
            bf[ns].u[1] = *(const uint4*)(p + 16);   // second 16B chunk at +32B
        }

        frag af[4];
#pragma unroll
        for (int ms = 0; ms < 4; ++ms) {
            const unsigned short* p = aptr[ms] + kc;
            af[ms].u[0] = *(const uint4*)(p);
            af[ms].u[1] = *(const uint4*)(p + 16);   // second 16B chunk at +32B
        }

#pragma unroll
        for (int ms = 0; ms < 4; ++ms)
#pragma unroll
            for (int ns = 0; ns < 4; ++ns)
                acc[ms][ns] = __builtin_amdgcn_wmma_f32_16x16x32_bf16(
                    false, af[ms].v, false, bf[ns].v,
                    (short)0, acc[ms][ns], false, false);
    }

    // Epilogue: clip and store fp32.
    // C/D layout: lanes 0-15 -> N=lane, M=e; lanes 16-31 -> N=lane-16, M=8+e.
#pragma unroll
    for (int ms = 0; ms < 4; ++ms)
#pragma unroll
        for (int ns = 0; ns < 4; ++ns) {
            const int m0 = bBase + ms * 16 + half * 8;
            const int n0 = oBase + ns * 16 + nlo;
#pragma unroll
            for (int e = 0; e < 8; ++e) {
                float v = acc[ms][ns][e];
                v = fminf(1.0f, fmaxf(-1.0f, v));
                out[(size_t)(m0 + e) * OUTF + n0] = v;
            }
        }
}

// ---------------------------------------------------------------------------
extern "C" void kernel_launch(void* const* d_in, const int* in_sizes, int n_in,
                              void* d_out, int out_size, void* d_ws, size_t ws_size,
                              hipStream_t stream) {
    const float* x  = (const float*)d_in[0];  // [8192,1024]
    const float* bw = (const float*)d_in[1];  // [1024,1024]
    const float* sw = (const float*)d_in[2];  // [1024,1024,11]
    float* out = (float*)d_out;               // [8192,1024]

    // Workspace layout: Wp bf16 [1024][12288] then Ax bf16 [8192][12288].
    unsigned short* wp = (unsigned short*)d_ws;                       // 25.2 MB
    unsigned short* ax = wp + (size_t)OUTF * KTOT;                    // +192 MB

    dim3 pgrid(KTOT / 256, OUTF);
    pack_weights_kernel<<<pgrid, 256, 0, stream>>>(bw, sw, wp);

    dim3 agrid(INF / 256, BATCH);
    build_A_kernel<<<agrid, 256, 0, stream>>>(x, ax);

    dim3 ggrid(OUTF / BN, BATCH / BM);
    kan_gemm_kernel<<<ggrid, 256, 0, stream>>>(ax, wp, out);
}
